// AttentionNoCache_19241453486811
// MI455X (gfx1250) — matrix operand order repaired and checked by reading, compiled, not run on hardware
//
#include <hip/hip_runtime.h>
#include <hip/hip_bf16.h>

#define BDIM 4
#define HDIM 16
#define LDIM 2048
#define DDIM 64
#define TOPKK 128
#define QT 16
#define SCALE 0.125f   // 64^-0.5

typedef __attribute__((ext_vector_type(16))) __bf16 v16bf;
typedef __attribute__((ext_vector_type(8)))  float  v8f;

__device__ __forceinline__ float wave_fmax(float v) {
#pragma unroll
    for (int o = 16; o > 0; o >>= 1) v = fmaxf(v, __shfl_xor(v, o, 32));
    return v;
}
__device__ __forceinline__ float wave_fsum(float v) {
#pragma unroll
    for (int o = 16; o > 0; o >>= 1) v += __shfl_xor(v, o, 32);
    return v;
}
// Wait for this wave's outstanding LDS ops + compiler memory barrier.
// DS ops are in-order per wave on CDNA5, so this makes prior lane-parallel
// LDS atomics/stores visible to subsequent loads from the same wave.
__device__ __forceinline__ void wave_lds_fence() {
    asm volatile("s_wait_dscnt 0x0" ::: "memory");
}
// Monotone float <-> uint32 ordering maps (larger float <=> larger key).
__device__ __forceinline__ unsigned f2k(float f) {
    unsigned u = __float_as_uint(f);
    return (u & 0x80000000u) ? ~u : (u | 0x80000000u);
}
__device__ __forceinline__ float k2f(unsigned k) {
    unsigned u = (k & 0x80000000u) ? (k & 0x7fffffffu) : ~k;
    return __uint_as_float(u);
}

// One workgroup (128 threads = 4 wave32) per 16-query tile of one (b,h).
__global__ __launch_bounds__(128) void topk_attn_kernel(
    const float* __restrict__ Q, const float* __restrict__ K,
    const float* __restrict__ V, float* __restrict__ out)
{
    extern __shared__ unsigned char smem_raw[];
    float*          scores = (float*)smem_raw;                       // 16*2048 f32 = 128KB
    unsigned*       hist   = (unsigned*)(smem_raw + 131072);         // 4 waves * 256 bins
    unsigned short* rowidx = (unsigned short*)(smem_raw + 135168);   // 16 * 256 u16
    float*          rowmax = (float*)(smem_raw + 143360);            // 16
    float*          rowthr = rowmax + 16;                            // 16
    float*          rowsum = rowthr + 16;                            // 16
    unsigned*       rowcnt = (unsigned*)(rowsum + 16);               // 16
    unsigned*       bcast  = rowcnt + 16;                            // 4 waves * 2

    const int tid  = threadIdx.x;
    const int lane = tid & 31;
    const int wave = tid >> 5;
    const int half = lane >> 4;
    const int mn   = lane & 15;

    const int qt = blockIdx.x & (LDIM / QT - 1);
    const int h  = (blockIdx.x >> 7) & (HDIM - 1);
    const int b  =  blockIdx.x >> 11;
    const int q0 = qt * QT;
    const int nkt = qt + 1;          // causal: only key tiles <= query tile
    const int nk  = nkt * QT;

    const size_t bh = ((size_t)b * HDIM + h) * LDIM;

    // ---------------- Phase 1: S = Q K^T (bf16 WMMA, f32 accumulate) -------
    const float* qrow = Q + (bh + q0 + mn) * DDIM;
    v16bf a0, a1;
#pragma unroll
    for (int i = 0; i < 8; ++i) {
        a0[i]     = (__bf16)qrow[      half * 16 + i];
        a0[i + 8] = (__bf16)qrow[8 +  half * 16 + i];
        a1[i]     = (__bf16)qrow[32 +  half * 16 + i];
        a1[i + 8] = (__bf16)qrow[40 +  half * 16 + i];
    }

    for (int kt = wave; kt < nkt; kt += 4) {
        const float* krow = K + (bh + kt * QT + mn) * DDIM;
        __builtin_prefetch(krow + 4 * QT * DDIM, 0, 0);
        v16bf b0, b1;
#pragma unroll
        for (int i = 0; i < 16; ++i) {
            b0[i] = (__bf16)krow[     half * 16 + i];
            b1[i] = (__bf16)krow[32 + half * 16 + i];
        }
        v8f c = {};
        c = __builtin_amdgcn_wmma_f32_16x16x32_bf16(false, a0, false, b0,
                                                    (short)0, c, false, false);
        c = __builtin_amdgcn_wmma_f32_16x16x32_bf16(false, a1, false, b1,
                                                    (short)0, c, false, false);
        const int kcol = kt * QT + mn;
#pragma unroll
        for (int r = 0; r < 8; ++r) {
            const int m = r + half * 8;
            float s = c[r];
            if (kcol > q0 + m) s = -3.0e38f;   // causal mask
            scores[m * LDIM + kcol] = s;
        }
    }
    __syncthreads();

    // ---------------- Phase 2: exact top-128 threshold via radix select ----
    // Block-uniform: every row in this tile needs selection iff nk > 128.
    const bool needSelect = (nk > TOPKK);
    unsigned* myhist = hist + wave * 256;

#pragma unroll 1
    for (int rl = 0; rl < 4; ++rl) {
        const int m  = wave * 4 + rl;
        const int rb = m * LDIM;

        float mx = -3.0e38f;
        for (int k = lane; k < nk; k += 32) mx = fmaxf(mx, scores[rb + k]);
        mx = wave_fmax(mx);

        float t = -1.0e37f;                 // default: keep all valid entries
        if (needSelect) {
            unsigned prefix = 0u;
            int rem = TOPKK;
#pragma unroll 1
            for (int pass = 0; pass < 4; ++pass) {
                const int shift = 24 - pass * 8;
                // zero this wave's 256-bin histogram (8 bins per lane)
#pragma unroll
                for (int i = 0; i < 8; ++i) myhist[lane + i * 32] = 0u;
                wave_lds_fence();
                // histogram digits of keys matching the decided prefix
                for (int k = lane; k < nk; k += 32) {
                    const unsigned key = f2k(scores[rb + k]);
                    if ((((unsigned long long)(key ^ prefix)) >> (shift + 8)) == 0ull)
                        atomicAdd(&myhist[(key >> shift) & 255u], 1u);
                }
                wave_lds_fence();
                // lane j owns bins [8j .. 8j+7]; scan buckets from the top
                unsigned hloc[8];
                unsigned ls = 0u;
#pragma unroll
                for (int i = 0; i < 8; ++i) { hloc[i] = myhist[lane * 8 + i]; ls += hloc[i]; }
                unsigned suf = ls;          // inclusive suffix-sum over lanes
#pragma unroll
                for (int o = 1; o < 32; o <<= 1) {
                    unsigned tval = __shfl_down(suf, o, 32);
                    if (lane + o < 32) suf += tval;
                }
                const unsigned above = suf - ls;   // count in strictly higher chunks
                const bool found = (above < (unsigned)rem) && ((unsigned)rem <= suf);
                if (found) {
                    unsigned cum = above;
                    for (int bb = 7; bb >= 0; --bb) {
                        cum += hloc[bb];
                        if (cum >= (unsigned)rem) {
                            bcast[wave * 2 + 0] = (unsigned)(lane * 8 + bb);
                            bcast[wave * 2 + 1] = (unsigned)(rem - (int)(cum - hloc[bb]));
                            break;
                        }
                    }
                }
                wave_lds_fence();
                const unsigned d = bcast[wave * 2 + 0];
                rem = (int)bcast[wave * 2 + 1];
                prefix |= (d << shift);
            }
            t = k2f(prefix);                // exact value of the 128th largest
        }

        // ---- compaction of selected keys + softmax denominator ----
        if (lane == 0) rowcnt[m] = 0u;
        wave_lds_fence();
        float se = 0.f;
        for (int k = lane; k < nk; k += 32) {
            const float s = scores[rb + k];
            if (s >= t) {
                const unsigned pos = atomicAdd(&rowcnt[m], 1u);
                if (pos < 256u) rowidx[m * 256 + pos] = (unsigned short)k;
                se += __expf((s - mx) * SCALE);
            }
        }
        se = wave_fsum(se);
        if (lane == 0) { rowmax[m] = mx; rowthr[m] = t; rowsum[m] = se; }
    }
    __syncthreads();

    // ---------------- Phase 3: out = attn @ V over the compacted keys ------
    {
        const int m  = tid >> 3;
        const int d0 = (tid & 7) * 8;
        const int rb = m * LDIM;
        const float mx  = rowmax[m];
        const float inv = 1.0f / rowsum[m];
        unsigned n = rowcnt[m];
        if (n > 256u) n = 256u;

        float acc[8];
#pragma unroll
        for (int j = 0; j < 8; ++j) acc[j] = 0.f;

        const float* vbase = V + bh * DDIM;
        for (unsigned i = 0; i < n; ++i) {
            const int k = (int)rowidx[m * 256 + i];
            const float s = scores[rb + k];
            const float w = __expf((s - mx) * SCALE) * inv;
            const float* vrow = vbase + (size_t)k * DDIM + d0;
#pragma unroll
            for (int j = 0; j < 8; ++j) acc[j] += w * vrow[j];
        }
        float* orow = out + (bh + q0 + m) * DDIM + d0;
#pragma unroll
        for (int j = 0; j < 8; ++j) orow[j] = acc[j];
    }
}

extern "C" void kernel_launch(void* const* d_in, const int* in_sizes, int n_in,
                              void* d_out, int out_size, void* d_ws, size_t ws_size,
                              hipStream_t stream) {
    (void)in_sizes; (void)n_in; (void)out_size; (void)d_ws; (void)ws_size;
    const float* Q = (const float*)d_in[0];
    const float* K = (const float*)d_in[1];
    const float* V = (const float*)d_in[2];
    float* out = (float*)d_out;

    const size_t shmem = 143872;   // scores + hist + idx + aux (~140.5 KB)
    hipFuncSetAttribute((const void*)topk_attn_kernel,
                        hipFuncAttributeMaxDynamicSharedMemorySize, (int)shmem);

    dim3 grid(BDIM * HDIM * (LDIM / QT));   // 8192 workgroups
    dim3 block(128);                        // 4 wave32
    topk_attn_kernel<<<grid, block, shmem, stream>>>(Q, K, V, out);
}
